// CausalSelfAttention_28028956574308
// MI455X (gfx1250) — matrix operand order repaired and checked by reading, not compiled, not run on hardware
//
#include <hip/hip_runtime.h>

#define BDIM 2
#define TSEQ 2048
#define CDIM 1024
#define HN 16
#define HD 64

typedef __attribute__((ext_vector_type(16))) __bf16 bf16x16;
typedef __attribute__((ext_vector_type(8)))  float  f32x8;
typedef __attribute__((ext_vector_type(4)))  unsigned int u32x4;

union Frag {
  u32x4   u[2];
  bf16x16 v;
};

__device__ __forceinline__ unsigned short f2bf(float f) {
  union { float f; unsigned u; } x; x.f = f;
  unsigned r = x.u + 0x7FFFu + ((x.u >> 16) & 1u);  // round-to-nearest-even
  return (unsigned short)(r >> 16);
}

// Async global -> LDS copy of 16 bytes (CDNA5 GLOBAL_LOAD_ASYNC_TO_LDS_B128).
// LDS destination address = low 32 bits of the generic pointer (aperture map).
__device__ __forceinline__ void g2lds_b128(unsigned short* lds_dst,
                                           const unsigned short* gsrc) {
  unsigned l = (unsigned)(size_t)lds_dst;
  asm volatile("global_load_async_to_lds_b128 %0, %1, off"
               :: "v"(l), "v"(gsrc) : "memory");
}
__device__ __forceinline__ void wait_async() {
  asm volatile("s_wait_asynccnt 0x0" ::: "memory");
}

// ---------------------------------------------------------------------------
// f32 -> bf16 elementwise conversion (verbatim layout)
// ---------------------------------------------------------------------------
__global__ void cvt_f32_bf16(const float* __restrict__ in,
                             unsigned short* __restrict__ out, int n) {
  int i = blockIdx.x * blockDim.x + threadIdx.x;
  if (i < n) out[i] = f2bf(in[i]);
}

// f32 [K][N] -> bf16 transposed [N][K] (one-shot weight transpose)
__global__ void cvt_t_f32_bf16(const float* __restrict__ in,
                               unsigned short* __restrict__ out, int K, int N) {
  int i = blockIdx.x * blockDim.x + threadIdx.x;  // k fastest -> coalesced writes
  if (i < K * N) {
    int k = i % K, n = i / K;
    out[i] = f2bf(in[(size_t)k * N + n]);
  }
}

// Rewrite V slice of qkv (bf16 [B*T][3C], cols 2C..3C) into vt [B][H][D][T]
__global__ void vtrans_bf16(const unsigned short* __restrict__ qkvb,
                            unsigned short* __restrict__ vt) {
  size_t i = (size_t)blockIdx.x * blockDim.x + threadIdx.x;  // t fastest
  int tt = (int)(i % TSEQ);
  size_t rest = i / TSEQ;
  int d = (int)(rest % HD); rest /= HD;
  int h = (int)(rest % HN);
  int b = (int)(rest / HN);
  vt[i] = qkvb[(size_t)(b * TSEQ + tt) * (3 * CDIM) + 2 * CDIM + h * HD + d];
}

// ---------------------------------------------------------------------------
// Tiled bf16 WMMA GEMM: C[M,N] = A[M,K] * Bt[N,K]^T + bias[N]
// 256 threads = 8 waves; block tile 128(M) x 64(N); K stepped by 32.
// All LDS staging via async b128 copies (no VGPR round-trip, no transposes).
// ---------------------------------------------------------------------------
template <int OUTF32>
__global__ __launch_bounds__(256) void gemm_wmma_bf16(
    const unsigned short* __restrict__ A,   // [M][K]
    const unsigned short* __restrict__ Bt,  // [N][K] (pre-transposed)
    const float* __restrict__ bias, void* __restrict__ Cout,
    int M, int N, int K) {
  __shared__ __align__(16) unsigned short As[128 * 32];   // [row][k]
  __shared__ __align__(16) unsigned short Bts[64 * 32];   // [n][k]

  const int t = threadIdx.x;
  const int lane = t & 31, w = t >> 5;
  const int half = lane >> 4, l16 = lane & 15;
  const int m0 = blockIdx.x * 128, n0 = blockIdx.y * 64;
  const int srow = t >> 1, scol = (t & 1) << 4;  // staging chunk: 2 chunks/row

  f32x8 acc[4] = {};

  for (int k0 = 0; k0 < K; k0 += 32) {
    // ---- async stage A (128x32) and Bt (64x32) tiles, verbatim rows ----
    g2lds_b128(As + srow * 32 + scol, A + (size_t)(m0 + srow) * K + k0 + scol);
    if (t < 128)
      g2lds_b128(Bts + srow * 32 + scol, Bt + (size_t)(n0 + srow) * K + k0 + scol);
    wait_async();
    __syncthreads();

    // A fragment: lane holds row (w*16 + l16); element i -> k = i + 16*half
    Frag a;
    const unsigned short* ar = As + (w * 16 + l16) * 32;
    a.u[0] = *(const u32x4*)(ar + 16 * half);
    a.u[1] = *(const u32x4*)(ar + 16 * half + 8);

#pragma unroll
    for (int nt = 0; nt < 4; ++nt) {
      // B fragment: lane holds col (nt*16 + l16); element i -> k = i + 16*half
      Frag b;
      const unsigned short* br = Bts + (nt * 16 + l16) * 32 + 16 * half;
      b.u[0] = *(const u32x4*)(br);
      b.u[1] = *(const u32x4*)(br + 8);
      acc[nt] = __builtin_amdgcn_wmma_f32_16x16x32_bf16(
          false, a.v, false, b.v, (short)0, acc[nt], false, false);
    }
    __syncthreads();
  }

  // ---- epilogue: bias add + store (C layout: elem r -> M = r + 8*half, N = l16) ----
#pragma unroll
  for (int nt = 0; nt < 4; ++nt) {
    int col = n0 + nt * 16 + l16;
    float bv = bias[col];
#pragma unroll
    for (int r = 0; r < 8; ++r) {
      int row = m0 + w * 16 + 8 * half + r;
      float v = acc[nt][r] + bv;
      if (OUTF32)
        ((float*)Cout)[(size_t)row * N + col] = v;
      else
        ((unsigned short*)Cout)[(size_t)row * N + col] = f2bf(v);
    }
  }
}

// ---------------------------------------------------------------------------
// Flash-style causal attention.
// grid.x = B*H, grid.y = T/64; 128 threads = 4 waves, each wave owns 16 q-rows.
// qkv: bf16 [B*T][3C]; vt: bf16 [B][H][D][T]; out yb: bf16 [B*T][C]
// ---------------------------------------------------------------------------
__global__ __launch_bounds__(128) void attn_wmma(
    const unsigned short* __restrict__ qkv,
    const unsigned short* __restrict__ vt,
    unsigned short* __restrict__ yb) {
  const int bh = blockIdx.x;
  const int b = bh / HN, h = bh % HN;
  const int qt = blockIdx.y;
  const int q0 = qt * 64;
  const int t = threadIdx.x;
  const int lane = t & 31, w = t >> 5;
  const int half = lane >> 4, l16 = lane & 15;

  __shared__ __align__(16) unsigned short Qs[64 * 64];    // [qrow][d]
  __shared__ __align__(16) unsigned short Ks[64 * 64];    // [key][d]  (Bt for Q*K^T)
  __shared__ __align__(16) unsigned short Vts[64 * 64];   // [d][key]  (Bt for P*V)
  __shared__ __align__(16) unsigned short Ps[4][16 * 64]; // per-wave P, [qrow][key]

  const size_t rs = 3 * CDIM;  // qkv row stride (elements)
  const unsigned short* qb = qkv + (size_t)(b * TSEQ + q0) * rs + h * HD;

  // ---- async stage Q tile 64x64 (8 b128 chunks per row) ----
#pragma unroll
  for (int i = 0; i < 4; ++i) {
    int idx = t * 4 + i;                  // 0..511
    int r = idx >> 3, c = (idx & 7) << 3;
    g2lds_b128(Qs + r * 64 + c, qb + (size_t)r * rs + c);
  }

  const float NEG_BIG = -3.0e38f;
  float m[8], l[8];
#pragma unroll
  for (int r = 0; r < 8; ++r) { m[r] = NEG_BIG; l[r] = 0.0f; }
  f32x8 o[4] = {};

  wait_async();
  __syncthreads();

  for (int j = 0; j <= qt; ++j) {
    // ---- async stage K [key][d] and Vt [d][key], both verbatim row copies ----
    const unsigned short* kb =
        qkv + (size_t)(b * TSEQ + j * 64) * rs + CDIM + h * HD;
    const unsigned short* vb =
        vt + (size_t)(b * HN + h) * HD * TSEQ + j * 64;
#pragma unroll
    for (int i = 0; i < 4; ++i) {
      int idx = t * 4 + i;
      int r = idx >> 3, c = (idx & 7) << 3;
      g2lds_b128(Ks + r * 64 + c, kb + (size_t)r * rs + c);
      g2lds_b128(Vts + r * 64 + c, vb + (size_t)r * TSEQ + c);
    }
    wait_async();
    __syncthreads();

    // ---- S = Q * K^T (16 q-rows x 64 keys per wave) ----
    f32x8 s[4] = {};
#pragma unroll
    for (int nt = 0; nt < 4; ++nt) {
#pragma unroll
      for (int ks = 0; ks < 2; ++ks) {
        Frag a, bf;
        const unsigned short* ar = Qs + (w * 16 + l16) * 64 + 32 * ks;
        a.u[0] = *(const u32x4*)(ar + 16 * half);
        a.u[1] = *(const u32x4*)(ar + 16 * half + 8);
        const unsigned short* br = Ks + (nt * 16 + l16) * 64 + 32 * ks + 16 * half;
        bf.u[0] = *(const u32x4*)(br);
        bf.u[1] = *(const u32x4*)(br + 8);
        s[nt] = __builtin_amdgcn_wmma_f32_16x16x32_bf16(
            false, a.v, false, bf.v, (short)0, s[nt], false, false);
      }
    }

    // ---- online softmax in C-register layout ----
    const float scale = 0.125f;  // 1/sqrt(64)
    const int qrow_base = q0 + w * 16 + 8 * half;  // + r
    const int kcol_base = j * 64 + l16;            // + nt*16
#pragma unroll
    for (int r = 0; r < 8; ++r) {
      float mx = NEG_BIG;
#pragma unroll
      for (int nt = 0; nt < 4; ++nt) {
        float v = s[nt][r] * scale;
        bool ok = (kcol_base + nt * 16) <= (qrow_base + r);  // causal
        v = ok ? v : NEG_BIG;
        s[nt][r] = v;
        mx = fmaxf(mx, v);
      }
#pragma unroll
      for (int off = 1; off <= 8; off <<= 1)
        mx = fmaxf(mx, __shfl_xor(mx, off));  // reduce over the 16-lane half
      float mnew = fmaxf(m[r], mx);
      float alpha = __expf(m[r] - mnew);
      m[r] = mnew;
      float rsum = 0.0f;
#pragma unroll
      for (int nt = 0; nt < 4; ++nt) {
        float p = __expf(s[nt][r] - mnew);
        rsum += p;
        Ps[w][(r + 8 * half) * 64 + nt * 16 + l16] = f2bf(p);
      }
#pragma unroll
      for (int off = 1; off <= 8; off <<= 1) rsum += __shfl_xor(rsum, off);
      l[r] = l[r] * alpha + rsum;
#pragma unroll
      for (int nt = 0; nt < 4; ++nt) o[nt][r] *= alpha;
    }

    // ---- O += P * V ----
#pragma unroll
    for (int nt = 0; nt < 4; ++nt) {
#pragma unroll
      for (int ks = 0; ks < 2; ++ks) {
        Frag a, bf;
        const unsigned short* ar = Ps[w] + l16 * 64 + 32 * ks;
        a.u[0] = *(const u32x4*)(ar + 16 * half);
        a.u[1] = *(const u32x4*)(ar + 16 * half + 8);
        const unsigned short* br = Vts + (nt * 16 + l16) * 64 + 32 * ks + 16 * half;
        bf.u[0] = *(const u32x4*)(br);
        bf.u[1] = *(const u32x4*)(br + 8);
        o[nt] = __builtin_amdgcn_wmma_f32_16x16x32_bf16(
            false, a.v, false, bf.v, (short)0, o[nt], false, false);
      }
    }
    __syncthreads();
  }

  // ---- normalize and store y (bf16, head h columns) ----
#pragma unroll
  for (int nt = 0; nt < 4; ++nt) {
#pragma unroll
    for (int r = 0; r < 8; ++r) {
      int row = q0 + w * 16 + 8 * half + r;
      int col = h * HD + nt * 16 + l16;
      float v = o[nt][r] / l[r];
      yb[(size_t)(b * TSEQ + row) * CDIM + col] = f2bf(v);
    }
  }
}

// ---------------------------------------------------------------------------
// Launch
// ---------------------------------------------------------------------------
extern "C" void kernel_launch(void* const* d_in, const int* in_sizes, int n_in,
                              void* d_out, int out_size, void* d_ws, size_t ws_size,
                              hipStream_t stream) {
  (void)in_sizes; (void)n_in; (void)out_size; (void)ws_size;

  const float* x      = (const float*)d_in[0];
  const float* W_attn = (const float*)d_in[1];
  const float* b_attn = (const float*)d_in[2];
  const float* W_proj = (const float*)d_in[3];
  const float* b_proj = (const float*)d_in[4];
  float* out = (float*)d_out;

  const int M  = BDIM * TSEQ;  // 4096
  const int K  = CDIM;         // 1024
  const int N1 = 3 * CDIM;     // 3072

  // workspace layout (bf16 elements): ~56 MB total
  unsigned short* xb   = (unsigned short*)d_ws;
  unsigned short* Wat  = xb   + (size_t)M * K;      // x bf16            (8 MB)
  unsigned short* Wpt  = Wat  + (size_t)N1 * K;     // W_attn^T [3C][C]  (6 MB)
  unsigned short* qkvb = Wpt  + (size_t)CDIM * K;   // W_proj^T [C][C]   (2 MB)
  unsigned short* vtb  = qkvb + (size_t)M * N1;     // qkv bf16          (24 MB)
  unsigned short* yb   = vtb  + (size_t)M * CDIM;   // V^T [B][H][D][T]  (8 MB)
                                                    // y bf16            (8 MB)

  int n;
  n = M * K;
  cvt_f32_bf16<<<(n + 255) / 256, 256, 0, stream>>>(x, xb, n);
  n = K * N1;
  cvt_t_f32_bf16<<<(n + 255) / 256, 256, 0, stream>>>(W_attn, Wat, K, N1);
  n = K * CDIM;
  cvt_t_f32_bf16<<<(n + 255) / 256, 256, 0, stream>>>(W_proj, Wpt, K, CDIM);

  // qkv = x @ W_attn + b_attn   (bf16 out)
  gemm_wmma_bf16<0><<<dim3(M / 128, N1 / 64), 256, 0, stream>>>(
      xb, Wat, b_attn, qkvb, M, N1, K);

  // one-shot V transpose into [B][H][D][T]
  n = M * CDIM;
  vtrans_bf16<<<(n + 255) / 256, 256, 0, stream>>>(qkvb, vtb);

  // attention -> yb (bf16)
  attn_wmma<<<dim3(BDIM * HN, TSEQ / 64), 128, 0, stream>>>(qkvb, vtb, yb);

  // out = yb @ W_proj + b_proj  (f32 out)
  gemm_wmma_bf16<1><<<dim3(M / 128, CDIM / 64), 256, 0, stream>>>(
      yb, Wpt, b_proj, out, M, CDIM, K);
}
